// TriplaneRenderer_20435454394786
// MI455X (gfx1250) — matrix operand order repaired and checked by reading, not compiled, not run on hardware
//
#include <hip/hip_runtime.h>
#include <hip/hip_bf16.h>
#include <math.h>

typedef _Float16 f16;
typedef __attribute__((ext_vector_type(16))) _Float16 v16h;
typedef __attribute__((ext_vector_type(8)))  _Float16 v8h;
typedef __attribute__((ext_vector_type(8)))  float    v8f;

#define NBATCH 2
#define NH 64
#define NW 64
#define NS 24
#define CP 32
#define RR 256
#define NRAY (NBATCH*NH*NW)        /* 8192 */
#define NPTS (NRAY*NS)             /* 196608 */
#define VSTR 136                   /* f16 stride for 128-col activation tile */
#define H2STR 264                  /* f16 stride for 256-col h2 tile */
#define PI_F 3.14159265358979f
#define LOG2E_F 1.44269504088896f
#define LN2_F 0.69314718055995f

/* workspace byte offsets */
#define WS_SIG0   0                       /* 256x128 f16 */
#define WS_V0     (WS_SIG0 + 65536)       /* 256x128 f16 */
#define WS_V1     (WS_V0 + 65536)         /* 256x256 f16 */
#define WS_RGB2   (WS_V1 + 131072)        /* 3x256 f32  */
#define WS_BRGB2  (WS_RGB2 + 3072)        /* 3 f32 (pad 64) */
#define WS_WSUM   (WS_BRGB2 + 64)         /* 8192 f32 */
#define WS_RAYH2  (WS_WSUM + 32768)       /* 8192x256 f16 */

/* d_out float offsets: rgb_map, feature_map, sdf, mask, xyz */
#define O_RGB   0
#define O_FEAT  (NRAY*3)
#define O_SDF   (O_FEAT + NRAY*256)
#define O_MASK  (O_SDF + NPTS)
#define O_XYZ   (O_MASK + NRAY)

/* fast exp2/log2: raw TRANS-pipe ops in the device pass; host pass only needs
   these to parse (device fn bodies are type-checked on host too). */
#if defined(__HIP_DEVICE_COMPILE__)
# if __has_builtin(__builtin_amdgcn_exp2f)
#  define FAST_EXP2(x) __builtin_amdgcn_exp2f(x)
# else
#  define FAST_EXP2(x) exp2f(x)
# endif
# if __has_builtin(__builtin_amdgcn_logf)
#  define FAST_LOG2(x) __builtin_amdgcn_logf(x)
# else
#  define FAST_LOG2(x) __log2f(x)
# endif
#else
# define FAST_EXP2(x) exp2f(x)
# define FAST_LOG2(x) log2f(x)
#endif

__device__ __forceinline__ v16h mk16(v8h lo, v8h hi) {
  v16h r;
#pragma unroll
  for (int i = 0; i < 8; ++i) { r[i] = lo[i]; r[i + 8] = hi[i]; }
  return r;
}

__device__ __forceinline__ v8f wmma16(v16h a, v16h b, v8f c) {
  return __builtin_amdgcn_wmma_f32_16x16x32_f16(false, a, false, b, (short)0, c, false, false);
}

/* B fragment (32x16, K x N) from row-major (N x K) f16 weights.
   lanes 0-15: cols, K {0..7,16..23} ; lanes 16-31: K {8..15,24..31} */
__device__ __forceinline__ v16h ldB(const f16* __restrict__ w, int krow, int nt, int kt, int lane) {
  int n = nt * 16 + (lane & 15);
  int k = kt * 32 + ((lane & 16) ? 8 : 0);
  const v8h* p = (const v8h*)(w + n * krow + k);
  return mk16(p[0], p[2]);
}

/* A fragment (16x32) from row-major f16 (LDS or global).
   lanes 0-15: K {0..7,16..23}; lanes 16-31: K {8..15,24..31} */
__device__ __forceinline__ v16h ldA(const f16* __restrict__ base, int stride, int row0, int kt, int lane) {
  int row = row0 + (lane & 15);
  int k = kt * 32 + ((lane & 16) ? 8 : 0);
  const v8h* lo = (const v8h*)(base + row * stride + k);
  const v8h* hi = (const v8h*)(base + row * stride + k + 16);
  return mk16(*lo, *hi);
}

/* branch-free stable softplus on raw TRANS ops; log2 arg is in (1,2] so no
   denorm/range fixups are needed: max(x,0) + ln2 * log2(1 + 2^(-|x|*log2e)) */
__device__ __forceinline__ float softplusf(float x) {
  float t = FAST_EXP2(-fabsf(x) * LOG2E_F);
  return fmaxf(x, 0.f) + LN2_F * FAST_LOG2(1.f + t);
}
__device__ __forceinline__ float fsigmoid(float x) {
  return 1.f / (1.f + FAST_EXP2(-x * LOG2E_F));
}

__device__ __forceinline__ void sample32(const float* __restrict__ base, float cx, float cy,
                                         f16* __restrict__ dst) {
  float x = ((cx + 1.f) * 256.f - 1.f) * 0.5f;
  float y = ((cy + 1.f) * 256.f - 1.f) * 0.5f;
  float fx = floorf(x), fy = floorf(y);
  float wx = x - fx, wy = y - fy;
  int x0 = (int)fx, y0 = (int)fy, x1 = x0 + 1, y1 = y0 + 1;
  float vx0 = (x0 >= 0 && x0 < 256) ? 1.f : 0.f;
  float vx1 = (x1 >= 0 && x1 < 256) ? 1.f : 0.f;
  float vy0 = (y0 >= 0 && y0 < 256) ? 1.f : 0.f;
  float vy1 = (y1 >= 0 && y1 < 256) ? 1.f : 0.f;
  int cx0 = x0 < 0 ? 0 : (x0 > 255 ? 255 : x0);
  int cx1 = x1 < 0 ? 0 : (x1 > 255 ? 255 : x1);
  int cy0 = y0 < 0 ? 0 : (y0 > 255 ? 255 : y0);
  int cy1 = y1 < 0 ? 0 : (y1 > 255 ? 255 : y1);
  float w00 = (1.f - wx) * (1.f - wy) * vx0 * vy0;
  float w10 = wx * (1.f - wy) * vx1 * vy0;
  float w01 = (1.f - wx) * wy * vx0 * vy1;
  float w11 = wx * wy * vx1 * vy1;
  int i00 = cy0 * 256 + cx0, i10 = cy0 * 256 + cx1;
  int i01 = cy1 * 256 + cx0, i11 = cy1 * 256 + cx1;
#pragma unroll 4
  for (int c = 0; c < 32; ++c) {
    const float* pc = base + c * 65536;
    dst[c] = (f16)(w00 * pc[i00] + w10 * pc[i10] + w01 * pc[i01] + w11 * pc[i11]);
  }
}

/* ---------------- weight prep: f32 -> padded f16 + fold rgb head ---------------- */
__global__ void prep_weights(const float* __restrict__ w_sig0, const float* __restrict__ w_v0,
                             const float* __restrict__ w_v1, const float* __restrict__ w_rgb,
                             const float* __restrict__ b_v1, const float* __restrict__ b_rgb,
                             f16* __restrict__ w_sig0p, f16* __restrict__ w_v0p,
                             f16* __restrict__ w_v1h, float* __restrict__ w_rgb2,
                             float* __restrict__ b_rgb2) {
  int tid = blockIdx.x * blockDim.x + threadIdx.x;
  if (tid < 256 * 128) {
    int n = tid >> 7, k = tid & 127;
    w_sig0p[tid] = (k < 96) ? (f16)w_sig0[n * 96 + k] : (f16)0.f;
    w_v0p[tid] = (k < 123) ? (f16)w_v0[n * 123 + k] : (f16)0.f;
  }
  if (tid < 65536) w_v1h[tid] = (f16)w_v1[tid];
  if (tid < 768) {
    int c = tid >> 8, k = tid & 255;
    float s = 0.f;
    for (int j = 0; j < 256; ++j) s += w_rgb[c * 256 + j] * w_v1[j * 256 + k];
    w_rgb2[tid] = s;
    if (k == 0) {
      float b = b_rgb[c];
      for (int j = 0; j < 256; ++j) b += w_rgb[c * 256 + j] * b_v1[j];
      b_rgb2[c] = b;
    }
  }
}

/* ---------------- main fused kernel: 1 block = 2 rays = 48 pts = 3 waves ---------------- */
__global__ __launch_bounds__(96) void triplane_main(
    const float* __restrict__ planes, const float* __restrict__ pts,
    const float* __restrict__ rays_d, const float* __restrict__ viewdirs,
    const float* __restrict__ z_vals, const float* __restrict__ nearp,
    const float* __restrict__ farp, const float* __restrict__ b_sig0,
    const float* __restrict__ w_sig1, const float* __restrict__ b_sig1,
    const float* __restrict__ b_v0, const float* __restrict__ sigbeta,
    const f16* __restrict__ w_sig0p, const f16* __restrict__ w_v0p,
    const float* __restrict__ w_rgb2, const float* __restrict__ b_rgb2,
    float* __restrict__ out, float* __restrict__ wsum_out, f16* __restrict__ rayh2) {
  __shared__ __align__(16) f16 vt[48 * VSTR];
  __shared__ __align__(16) f16 h2[48 * H2STR];
  __shared__ float sdfs[48];
  __shared__ float rgbs[48 * 3];
  __shared__ float wgt[48];

  const int tid = threadIdx.x;
  const int lane = tid & 31;
  const int mb = (tid >> 5) * 16;     /* wave's local row base: 0/16/32 */
  const int P0 = blockIdx.x * 48;

  /* ---- stage 1: activations (triplane bilinear + pos-encode) -> vt ---- */
  {
    const int ptl = mb + (lane & 15);
    const int p = P0 + ptl;
    const float nr = nearp[0], fr = farp[0];
    float c3[3];
#pragma unroll
    for (int d = 0; d < 3; ++d) c3[d] = 2.f * (pts[p * 3 + d] - nr) / (fr - nr) - 1.f;
    const int bi = p / (NH * NW * NS);
    const size_t pbase = (size_t)bi * 3 * CP * (RR * RR);
    f16* row = &vt[ptl * VSTR];
    if ((lane & 16) == 0) {
      sample32(planes + pbase, c3[0], c3[1], row);                             /* plane 0: (x,y) */
      sample32(planes + pbase + (size_t)CP * RR * RR, c3[0], c3[2], row + 32); /* plane 1: (x,z) */
    } else {
      sample32(planes + pbase + (size_t)2 * CP * RR * RR, c3[1], c3[2], row + 64); /* plane 2 */
      const int ray = p / NS;
      float vd[3];
#pragma unroll
      for (int d = 0; d < 3; ++d) vd[d] = viewdirs[ray * 3 + d];
#pragma unroll
      for (int d = 0; d < 3; ++d) row[96 + d] = (f16)vd[d];
#pragma unroll
      for (int k = 0; k < 4; ++k) {
        float sc = (float)(1 << k) * PI_F;
#pragma unroll
        for (int d = 0; d < 3; ++d) {
          row[99 + k * 3 + d]  = (f16)__sinf(sc * vd[d]);
          row[111 + k * 3 + d] = (f16)__cosf(sc * vd[d]);
        }
      }
#pragma unroll
      for (int k = 123; k < 128; ++k) row[k] = (f16)0.f;
    }
  }
  /* intra-wave LDS producer/consumer: DS ops are in-order per wave */

  /* ---- A fragments for the 16x128 activation tile ---- */
  v16h av[4];
#pragma unroll
  for (int kt = 0; kt < 4; ++kt) av[kt] = ldA(vt, VSTR, mb, kt, lane);

  /* ---- GEMM1 (sig, K=128) with fused w_sig1 dot -> sdf ---- */
  float sacc[8];
#pragma unroll
  for (int g = 0; g < 8; ++g) sacc[g] = 0.f;
  for (int nt = 0; nt < 16; ++nt) {
    int col = nt * 16 + (lane & 15);
    float bias = b_sig0[col], w1 = w_sig1[col];
    v8f acc;
#pragma unroll
    for (int g = 0; g < 8; ++g) acc[g] = bias;   /* bias rides the MMA accumulator */
#pragma unroll
    for (int kt = 0; kt < 4; ++kt) acc = wmma16(av[kt], ldB(w_sig0p, 128, nt, kt, lane), acc);
#pragma unroll
    for (int g = 0; g < 8; ++g) sacc[g] += softplusf(acc[g]) * w1;
  }
#pragma unroll
  for (int g = 0; g < 8; ++g) {
    float s = sacc[g];
    s += __shfl_xor(s, 1); s += __shfl_xor(s, 2);
    s += __shfl_xor(s, 4); s += __shfl_xor(s, 8);
    sacc[g] = s;
  }
  if ((lane & 15) == 0) {
    float bs1 = b_sig1[0];
    int rbase = mb + ((lane & 16) ? 8 : 0);
#pragma unroll
    for (int g = 0; g < 8; ++g) {
      float sv = sacc[g] + bs1;
      sdfs[rbase + g] = sv;
      out[O_SDF + P0 + rbase + g] = sv;
    }
  }

  /* ---- GEMM2 (v0, K=128) -> softplus -> h2 (f16 in LDS) ---- */
  for (int nt = 0; nt < 16; ++nt) {
    int col = nt * 16 + (lane & 15);
    float bias = b_v0[col];
    v8f acc;
#pragma unroll
    for (int g = 0; g < 8; ++g) acc[g] = bias;
#pragma unroll
    for (int kt = 0; kt < 4; ++kt) acc = wmma16(av[kt], ldB(w_v0p, 128, nt, kt, lane), acc);
    int rbase = mb + ((lane & 16) ? 8 : 0);
#pragma unroll
    for (int g = 0; g < 8; ++g) h2[(rbase + g) * H2STR + col] = (f16)softplusf(acc[g]);
  }

  /* ---- per-point rgb from folded (W_rgb · W_v1) head ---- */
  {
    int ptl = mb + (lane & 15);
    int kb = (lane & 16) ? 128 : 0;
#pragma unroll
    for (int c = 0; c < 3; ++c) {
      float s = 0.f;
      for (int k = kb; k < kb + 128; k += 8) {
        v8h hv = *(const v8h*)(&h2[ptl * H2STR + k]);
#pragma unroll
        for (int i = 0; i < 8; ++i) s += (float)hv[i] * w_rgb2[c * 256 + k + i];
      }
      s += __shfl_xor(s, 16);
      if ((lane & 16) == 0) rgbs[ptl * 3 + c] = s + b_rgb2[c];
    }
  }

  __syncthreads();

  /* ---- volume render: 2 rays, sequential cumprod over N=24 ---- */
  if (tid < 2) {
    int r = tid;
    int rayG = blockIdx.x * 2 + r;
    float rd0 = rays_d[rayG * 3], rd1 = rays_d[rayG * 3 + 1], rd2 = rays_d[rayG * 3 + 2];
    float nrm = sqrtf(rd0 * rd0 + rd1 * rd1 + rd2 * rd2);
    float beta = sigbeta[0];
    float trans = 1.f, wsum = 0.f, lastw = 0.f;
    float arc[3] = {0.f, 0.f, 0.f}, axyz[3] = {0.f, 0.f, 0.f};
    for (int n = 0; n < NS; ++n) {
      float zn = z_vals[rayG * NS + n];
      float dist = ((n < NS - 1) ? (z_vals[rayG * NS + n + 1] - zn) : 1e10f) * nrm;
      float sd = sdfs[r * NS + n];
      float sigma = fsigmoid(-sd / beta) / beta;
      float alpha = 1.f - FAST_EXP2(-sigma * dist * LOG2E_F);
      float wn = alpha * trans;
      trans *= (1.f - alpha + 1e-10f);
      wgt[r * NS + n] = wn;
      wsum += wn;
      int pg = P0 + r * NS + n;
#pragma unroll
      for (int c = 0; c < 3; ++c) {
        float rg = rgbs[(r * NS + n) * 3 + c];
        arc[c] += wn * fsigmoid(rg);
        axyz[c] += wn * pts[pg * 3 + c];
      }
      lastw = wn;
    }
#pragma unroll
    for (int c = 0; c < 3; ++c) {
      out[O_RGB + rayG * 3 + c] = -1.f + 2.f * arc[c];
      out[O_XYZ + rayG * 3 + c] = axyz[c];
    }
    out[O_MASK + rayG] = lastw;
    wsum_out[rayG] = wsum;
  }

  __syncthreads();

  /* ---- per-ray weighted h2 sums (feature_map is linear in h2) ---- */
  for (int idx = tid; idx < 512; idx += 96) {
    int r = idx >> 8, k = idx & 255;
    float s = 0.f;
#pragma unroll
    for (int n = 0; n < NS; ++n) s += wgt[r * NS + n] * (float)h2[(r * NS + n) * H2STR + k];
    rayh2[(blockIdx.x * 2 + r) * 256 + k] = (f16)s;
  }
}

/* ---------------- feature_map GEMM: (8192x256) @ (256x256)^T + Σw·b_v1 ---------------- */
__global__ __launch_bounds__(128) void feat_gemm(const f16* __restrict__ rayh2,
                                                 const f16* __restrict__ w_v1h,
                                                 const float* __restrict__ b_v1,
                                                 const float* __restrict__ wsum,
                                                 float* __restrict__ out) {
  int lane = threadIdx.x & 31;
  int t = blockIdx.x * 4 + (threadIdx.x >> 5);   /* M-tile 0..511 */
  int rbase = t * 16;
  v16h a[8];
  {
    int row = rbase + (lane & 15);
#pragma unroll
    for (int kt = 0; kt < 8; ++kt) {
      int k = kt * 32 + ((lane & 16) ? 8 : 0);
      const v8h* lo = (const v8h*)(rayh2 + row * 256 + k);
      const v8h* hi = (const v8h*)(rayh2 + row * 256 + k + 16);
      a[kt] = mk16(*lo, *hi);
    }
  }
  int rb2 = rbase + ((lane & 16) ? 8 : 0);
  float wsv[8];
#pragma unroll
  for (int g = 0; g < 8; ++g) wsv[g] = wsum[rb2 + g];
  for (int nt = 0; nt < 16; ++nt) {
    int col = nt * 16 + (lane & 15);
    float bv = b_v1[col];
    v8f acc;
#pragma unroll
    for (int g = 0; g < 8; ++g) acc[g] = wsv[g] * bv;  /* Σw·b_v1 rides the accumulator */
#pragma unroll
    for (int kt = 0; kt < 8; ++kt) acc = wmma16(a[kt], ldB(w_v1h, 256, nt, kt, lane), acc);
#pragma unroll
    for (int g = 0; g < 8; ++g) out[O_FEAT + (rb2 + g) * 256 + col] = acc[g];
  }
}

extern "C" void kernel_launch(void* const* d_in, const int* in_sizes, int n_in,
                              void* d_out, int out_size, void* d_ws, size_t ws_size,
                              hipStream_t stream) {
  const float* planes   = (const float*)d_in[0];
  const float* pts      = (const float*)d_in[1];
  const float* rays_d   = (const float*)d_in[2];
  const float* viewdirs = (const float*)d_in[3];
  const float* z_vals   = (const float*)d_in[4];
  const float* nearp    = (const float*)d_in[5];
  const float* farp     = (const float*)d_in[6];
  const float* w_sig0   = (const float*)d_in[7];
  const float* b_sig0   = (const float*)d_in[8];
  const float* w_sig1   = (const float*)d_in[9];
  const float* b_sig1   = (const float*)d_in[10];
  const float* w_v0     = (const float*)d_in[11];
  const float* b_v0     = (const float*)d_in[12];
  const float* w_v1     = (const float*)d_in[13];
  const float* b_v1     = (const float*)d_in[14];
  const float* w_rgb    = (const float*)d_in[15];
  const float* b_rgb    = (const float*)d_in[16];
  const float* sigbeta  = (const float*)d_in[17];

  char* ws = (char*)d_ws;
  f16*   w_sig0p = (f16*)(ws + WS_SIG0);
  f16*   w_v0p   = (f16*)(ws + WS_V0);
  f16*   w_v1h   = (f16*)(ws + WS_V1);
  float* w_rgb2  = (float*)(ws + WS_RGB2);
  float* b_rgb2  = (float*)(ws + WS_BRGB2);
  float* wsum    = (float*)(ws + WS_WSUM);
  f16*   rayh2   = (f16*)(ws + WS_RAYH2);
  float* out     = (float*)d_out;

  prep_weights<<<256, 256, 0, stream>>>(w_sig0, w_v0, w_v1, w_rgb, b_v1, b_rgb,
                                        w_sig0p, w_v0p, w_v1h, w_rgb2, b_rgb2);
  triplane_main<<<NRAY / 2, 96, 0, stream>>>(
      planes, pts, rays_d, viewdirs, z_vals, nearp, farp,
      b_sig0, w_sig1, b_sig1, b_v0, sigbeta,
      w_sig0p, w_v0p, w_rgb2, b_rgb2, out, wsum, rayh2);
  feat_gemm<<<NRAY / 16 / 4, 128, 0, stream>>>(rayh2, w_v1h, b_v1, wsum, out);
}
